// BertAttention_83846351552574
// MI455X (gfx1250) — matrix operand order repaired and checked by reading, not compiled, not run on hardware
//
#include <hip/hip_runtime.h>

// BERT attention (B=2, S=4096, H=768) for MI455X / gfx1250.
// All matmuls via V_WMMA_F32_16X16X32_F16 (f16 inputs, fp32 accumulate).
// The 67MB f16 probability matrix is materialized in workspace: it fits in
// the 192MB L2, so attention runs as three peak-rate WMMA GEMMs + a softmax.
// The k-loop is 2x unrolled with ping-pong register buffers (no copy chains)
// and hoisted per-lane base pointers (no per-iteration 64-bit address math).

typedef __attribute__((ext_vector_type(16))) _Float16 v16h;
typedef __attribute__((ext_vector_type(8)))  _Float16 v8h;
typedef __attribute__((ext_vector_type(8)))  float    v8f;

#define HDIM  768
#define SEQ   4096
#define NBAT  2
#define BS    (NBAT * SEQ)

// ---------------------------------------------------------------------------
// WMMA fragment loads from hoisted per-lane base pointers
// (wave32; contiguous k = 16*(lane/16) + s, same order for A and B).
// A lane base: &A[(m0 + lane%16)*lda + 16*(lane/16)]; frag = [k..k+15] contiguous
// B lane base: &Bm[(n0 + lane%16)*ldb + 16*(lane/16)]; frag = [k..k+15] contiguous
// ---------------------------------------------------------------------------
__device__ __forceinline__ v16h frag_a(const _Float16* __restrict__ p, int k) {
  return *(const v16h*)(p + k);
}
__device__ __forceinline__ v16h frag_b(const _Float16* __restrict__ p, int k) {
  return *(const v16h*)(p + k);
}

// ---------------------------------------------------------------------------
// Generic WMMA GEMM: acc[m][n] = sum_k A[m][k] * Bm[n][k]
// 256 threads / 8 waves (2 rows x 4 cols). Wave tile 64 x (16*NT).
// Block tile 128 x (64*NT).  NT=2 -> 128x128, NT=4 -> 128x256.
// Requires K % 64 == 0 (true for 768 and 4096).
// Epilogue modes:
//   0: f16 out = (acc + bias?) * scale            (Q/K projections, ctx)
//   1: f16 out transposed: out[n*ldo + m] = acc+b (V projection -> Vt)
//   2: f16 out = acc * mask[n]                    (masked scores)
//   3: f32 out = acc + bias[n] + resid[m*ldo+n]   (Wo1/Wo2 + residual)
// ---------------------------------------------------------------------------
template <int NT>
__global__ __launch_bounds__(256)
void gemm_wmma_kernel(const _Float16* __restrict__ A,  int lda, long strideA,
                      const _Float16* __restrict__ Bm, int ldb, long strideB,
                      const float* __restrict__ bias,
                      const float* __restrict__ extra, long strideE,
                      void* __restrict__ outv, int ldo, long strideO,
                      int K, float scale, int mode)
{
  const int bz = blockIdx.z;
  A  += (long)bz * strideA;
  Bm += (long)bz * strideB;
  const float* extraB = extra ? (extra + (long)bz * strideE) : nullptr;

  const int wave = threadIdx.x >> 5;
  const int lane = threadIdx.x & 31;
  const int half = lane >> 4;
  const int lm   = lane & 15;
  const int wm = blockIdx.y * 128 + (wave >> 2) * 64;             // wave M origin
  const int wn = blockIdx.x * (64 * NT) + (wave & 3) * (16 * NT); // wave N origin

  // Hoisted per-lane fragment base pointers.
  const _Float16* pa[4];
#pragma unroll
  for (int i = 0; i < 4; ++i)
    pa[i] = A + (long)(wm + 16 * i + lm) * lda + 16 * half;
  const _Float16* pb[NT];
#pragma unroll
  for (int j = 0; j < NT; ++j)
    pb[j] = Bm + (long)(wn + 16 * j + lm) * ldb + 16 * half;

  v8f c[4][NT] = {};

  // Ping-pong register double buffer; 2x-unrolled k-loop (K % 64 == 0).
  v16h a0[4], b0[NT], a1[4], b1[NT];
#pragma unroll
  for (int i = 0; i < 4; ++i) a0[i] = frag_a(pa[i], 0);
#pragma unroll
  for (int j = 0; j < NT; ++j) b0[j] = frag_b(pb[j], 0);

  for (int k0 = 0; k0 < K; k0 += 64) {
    const int k1 = k0 + 32;  // always valid since K % 64 == 0
#pragma unroll
    for (int i = 0; i < 4; ++i) a1[i] = frag_a(pa[i], k1);
#pragma unroll
    for (int j = 0; j < NT; ++j) b1[j] = frag_b(pb[j], k1);

#pragma unroll
    for (int i = 0; i < 4; ++i)
#pragma unroll
      for (int j = 0; j < NT; ++j)
        c[i][j] = __builtin_amdgcn_wmma_f32_16x16x32_f16(
            false, a0[i], false, b0[j], (short)0, c[i][j], false, false);

    const int k2 = (k1 + 32 < K) ? (k1 + 32) : 0;  // clamp final prefetch
#pragma unroll
    for (int i = 0; i < 4; ++i) a0[i] = frag_a(pa[i], k2);
#pragma unroll
    for (int j = 0; j < NT; ++j) b0[j] = frag_b(pb[j], k2);

#pragma unroll
    for (int i = 0; i < 4; ++i)
#pragma unroll
      for (int j = 0; j < NT; ++j)
        c[i][j] = __builtin_amdgcn_wmma_f32_16x16x32_f16(
            false, a1[i], false, b1[j], (short)0, c[i][j], false, false);
  }

  // D layout: element e <-> row e + 8*(lane>>4), col lane&15
  const int col  = lm;
  const int rofs = half * 8;
#pragma unroll
  for (int i = 0; i < 4; ++i) {
#pragma unroll
    for (int j = 0; j < NT; ++j) {
      const int nn = wn + 16 * j + col;
#pragma unroll
      for (int e = 0; e < 8; ++e) {
        const int mm = wm + 16 * i + rofs + e;
        float v = c[i][j][e];
        if (mode == 0) {
          if (bias) v += bias[nn];
          v *= scale;
          ((_Float16*)outv)[(long)bz * strideO + (long)mm * ldo + nn] = (_Float16)v;
        } else if (mode == 1) {
          v += bias[nn];
          ((_Float16*)outv)[(long)bz * strideO + (long)nn * ldo + mm] = (_Float16)v;
        } else if (mode == 2) {
          v *= extraB[nn];
          ((_Float16*)outv)[(long)bz * strideO + (long)mm * ldo + nn] = (_Float16)v;
        } else {
          v += bias[nn] + extraB[(long)mm * ldo + nn];
          ((float*)outv)[(long)bz * strideO + (long)mm * ldo + nn] = v;
        }
      }
    }
  }
}

// ---------------------------------------------------------------------------
// fp32 -> f16 cast
// ---------------------------------------------------------------------------
__global__ void f32_to_f16_kernel(const float* __restrict__ src,
                                  _Float16* __restrict__ dst, int n) {
  int i = blockIdx.x * blockDim.x + threadIdx.x;
  if (i < n) dst[i] = (_Float16)src[i];
}

// ---------------------------------------------------------------------------
// In-place softmax over rows of length SEQ (f16). One 256-thread block / row.
// ---------------------------------------------------------------------------
__global__ __launch_bounds__(256)
void softmax_rows_kernel(_Float16* __restrict__ P) {
  _Float16* p = P + (long)blockIdx.x * SEQ;
  const int t = threadIdx.x;

  float x[16];
  v8h h0 = *(const v8h*)(p + t * 16);
  v8h h1 = *(const v8h*)(p + t * 16 + 8);
#pragma unroll
  for (int i = 0; i < 8; ++i) { x[i] = (float)h0[i]; x[8 + i] = (float)h1[i]; }

  float mx = x[0];
#pragma unroll
  for (int i = 1; i < 16; ++i) mx = fmaxf(mx, x[i]);
#pragma unroll
  for (int off = 16; off > 0; off >>= 1) mx = fmaxf(mx, __shfl_xor(mx, off, 32));

  __shared__ float red_m[8], red_s[8];
  const int wid = t >> 5;
  if ((t & 31) == 0) red_m[wid] = mx;
  __syncthreads();
  float m2 = red_m[0];
#pragma unroll
  for (int w = 1; w < 8; ++w) m2 = fmaxf(m2, red_m[w]);

  float s = 0.f;
#pragma unroll
  for (int i = 0; i < 16; ++i) { x[i] = __expf(x[i] - m2); s += x[i]; }
#pragma unroll
  for (int off = 16; off > 0; off >>= 1) s += __shfl_xor(s, off, 32);
  if ((t & 31) == 0) red_s[wid] = s;
  __syncthreads();
  float tot = 0.f;
#pragma unroll
  for (int w = 0; w < 8; ++w) tot += red_s[w];
  const float inv = 1.0f / tot;

  v8h o0, o1;
#pragma unroll
  for (int i = 0; i < 8; ++i) {
    o0[i] = (_Float16)(x[i] * inv);
    o1[i] = (_Float16)(x[8 + i] * inv);
  }
  *(v8h*)(p + t * 16)     = o0;
  *(v8h*)(p + t * 16 + 8) = o1;
}

// ---------------------------------------------------------------------------
// LayerNorm over rows of HDIM fp32 (ddof=0, eps=1e-12). 256 thr, 3 elems each.
// Optionally writes fp32 (final output) and/or f16 (next GEMM operand).
// ---------------------------------------------------------------------------
__global__ __launch_bounds__(256)
void layernorm_kernel(const float* __restrict__ X, const float* __restrict__ g,
                      const float* __restrict__ b, float* __restrict__ out32,
                      _Float16* __restrict__ out16) {
  const float* x = X + (long)blockIdx.x * HDIM;
  const int t = threadIdx.x;
  const float v0 = x[t], v1 = x[t + 256], v2 = x[t + 512];
  float s = v0 + v1 + v2;
  float q = v0 * v0 + v1 * v1 + v2 * v2;
#pragma unroll
  for (int off = 16; off > 0; off >>= 1) {
    s += __shfl_xor(s, off, 32);
    q += __shfl_xor(q, off, 32);
  }
  __shared__ float rs[8], rq[8];
  const int wid = t >> 5;
  if ((t & 31) == 0) { rs[wid] = s; rq[wid] = q; }
  __syncthreads();
  float S = 0.f, Q = 0.f;
#pragma unroll
  for (int w = 0; w < 8; ++w) { S += rs[w]; Q += rq[w]; }
  const float mean = S * (1.0f / HDIM);
  const float var  = Q * (1.0f / HDIM) - mean * mean;
  const float r    = rsqrtf(var + 1e-12f);

  const float vv[3] = { v0, v1, v2 };
#pragma unroll
  for (int i = 0; i < 3; ++i) {
    const int cix = t + i * 256;
    const float y = (vv[i] - mean) * r * g[cix] + b[cix];
    if (out32) out32[(long)blockIdx.x * HDIM + cix] = y;
    if (out16) out16[(long)blockIdx.x * HDIM + cix] = (_Float16)y;
  }
}

// ---------------------------------------------------------------------------
// Host-side orchestration
// ---------------------------------------------------------------------------
extern "C" void kernel_launch(void* const* d_in, const int* in_sizes, int n_in,
                              void* d_out, int out_size, void* d_ws, size_t ws_size,
                              hipStream_t stream) {
  const float* x    = (const float*)d_in[0];
  const float* mask = (const float*)d_in[1];
  const float* Wq   = (const float*)d_in[2];
  const float* bq   = (const float*)d_in[3];
  const float* Wk   = (const float*)d_in[4];
  const float* bk   = (const float*)d_in[5];
  const float* Wv   = (const float*)d_in[6];
  const float* bv   = (const float*)d_in[7];
  const float* Wo1  = (const float*)d_in[8];
  const float* bo1  = (const float*)d_in[9];
  const float* g1   = (const float*)d_in[10];
  const float* be1  = (const float*)d_in[11];
  const float* Wo2  = (const float*)d_in[12];
  const float* bo2  = (const float*)d_in[13];
  const float* g2   = (const float*)d_in[14];
  const float* be2  = (const float*)d_in[15];
  float* out = (float*)d_out;

  // Workspace carve-up (256B aligned). Total ~174MB.
  char* ws = (char*)d_ws;
  size_t off = 0;
  auto take = [&](size_t bytes) -> char* {
    char* p = ws + off;
    off += (bytes + 255) & ~(size_t)255;
    return p;
  };
  _Float16* Xh   = (_Float16*)take((size_t)BS * HDIM * 2);
  _Float16* Wqh  = (_Float16*)take((size_t)HDIM * HDIM * 2);
  _Float16* Wkh  = (_Float16*)take((size_t)HDIM * HDIM * 2);
  _Float16* Wvh  = (_Float16*)take((size_t)HDIM * HDIM * 2);
  _Float16* Wo1h = (_Float16*)take((size_t)HDIM * HDIM * 2);
  _Float16* Wo2h = (_Float16*)take((size_t)HDIM * HDIM * 2);
  _Float16* Qh   = (_Float16*)take((size_t)BS * HDIM * 2);
  _Float16* Kh   = (_Float16*)take((size_t)BS * HDIM * 2);
  _Float16* Vt   = (_Float16*)take((size_t)HDIM * BS * 2);  // [h][b*S+s]
  _Float16* P    = (_Float16*)take((size_t)NBAT * SEQ * SEQ * 2);
  _Float16* Ctxh = (_Float16*)take((size_t)BS * HDIM * 2);
  float*    Pre  = (float*)take((size_t)BS * HDIM * 4);     // reused for both pre-LN
  _Float16* H1h  = (_Float16*)take((size_t)BS * HDIM * 2);
  (void)ws_size; (void)in_sizes; (void)n_in; (void)out_size;

  const dim3 blk(256);
  const float inv_sqrt_h = 0.03608439182435161f;  // 1/sqrt(768)

  // --- casts to f16 ---
  {
    int n = BS * HDIM;
    f32_to_f16_kernel<<<(n + 255) / 256, 256, 0, stream>>>(x, Xh, n);
    int nw = HDIM * HDIM;
    f32_to_f16_kernel<<<(nw + 255) / 256, 256, 0, stream>>>(Wq,  Wqh,  nw);
    f32_to_f16_kernel<<<(nw + 255) / 256, 256, 0, stream>>>(Wk,  Wkh,  nw);
    f32_to_f16_kernel<<<(nw + 255) / 256, 256, 0, stream>>>(Wv,  Wvh,  nw);
    f32_to_f16_kernel<<<(nw + 255) / 256, 256, 0, stream>>>(Wo1, Wo1h, nw);
    f32_to_f16_kernel<<<(nw + 255) / 256, 256, 0, stream>>>(Wo2, Wo2h, nw);
  }

  // --- Q = (X Wq^T + bq)/sqrt(H), K = X Wk^T + bk, V -> Vt (f16) ---
  {
    dim3 grid(HDIM / 128, BS / 128, 1);
    gemm_wmma_kernel<2><<<grid, blk, 0, stream>>>(
        Xh, HDIM, 0, Wqh, HDIM, 0, bq, nullptr, 0,
        Qh, HDIM, 0, HDIM, inv_sqrt_h, 0);
    gemm_wmma_kernel<2><<<grid, blk, 0, stream>>>(
        Xh, HDIM, 0, Wkh, HDIM, 0, bk, nullptr, 0,
        Kh, HDIM, 0, HDIM, 1.0f, 0);
    gemm_wmma_kernel<2><<<grid, blk, 0, stream>>>(
        Xh, HDIM, 0, Wvh, HDIM, 0, bv, nullptr, 0,
        Vt, BS, 0, HDIM, 1.0f, 1);
  }

  // --- scores: P[b][q][k] = (Q.K^T) * mask[b][k]  (f16, lives in L2) ---
  // Wide tiles (128x256): plenty of blocks (16x32x2), 2x A reuse per wave.
  {
    dim3 grid(SEQ / 256, SEQ / 128, NBAT);
    gemm_wmma_kernel<4><<<grid, blk, 0, stream>>>(
        Qh, HDIM, (long)SEQ * HDIM, Kh, HDIM, (long)SEQ * HDIM,
        nullptr, mask, (long)SEQ,
        P, SEQ, (long)SEQ * SEQ, HDIM, 1.0f, 2);
  }

  // --- softmax over keys, in place ---
  softmax_rows_kernel<<<BS, 256, 0, stream>>>(P);

  // --- ctx = P @ V  (f16 out) ---
  {
    dim3 grid(HDIM / 128, SEQ / 128, NBAT);
    gemm_wmma_kernel<2><<<grid, blk, 0, stream>>>(
        P, SEQ, (long)SEQ * SEQ, Vt, BS, (long)SEQ,
        nullptr, nullptr, 0,
        Ctxh, HDIM, (long)SEQ * HDIM, SEQ, 1.0f, 0);
  }

  // --- h1pre = ctx Wo1^T + bo1 + x  (f32), then LN1 -> f16 ---
  {
    dim3 grid(HDIM / 128, BS / 128, 1);
    gemm_wmma_kernel<2><<<grid, blk, 0, stream>>>(
        Ctxh, HDIM, 0, Wo1h, HDIM, 0, bo1, x, 0,
        Pre, HDIM, 0, HDIM, 1.0f, 3);
  }
  layernorm_kernel<<<BS, 256, 0, stream>>>(Pre, g1, be1, nullptr, H1h);

  // --- out_pre = h1 Wo2^T + bo2 + x (f32, reusing Pre), then LN2 -> d_out ---
  {
    dim3 grid(HDIM / 128, BS / 128, 1);
    gemm_wmma_kernel<2><<<grid, blk, 0, stream>>>(
        H1h, HDIM, 0, Wo2h, HDIM, 0, bo2, x, 0,
        Pre, HDIM, 0, HDIM, 1.0f, 3);
  }
  layernorm_kernel<<<BS, 256, 0, stream>>>(Pre, g2, be2, out, nullptr);
}
